// MutliHeadSparseAttention_66675072303641
// MI455X (gfx1250) — matrix operand order repaired and checked by reading, not compiled, not run on hardware
//
#include <hip/hip_runtime.h>

typedef __attribute__((ext_vector_type(16))) __bf16 v16bf;
typedef __attribute__((ext_vector_type(8)))  float  v8f;

#define D_MODEL 1024
#define N_HEAD  16
#define DH      64
#define SEQ     1024
#define BATCH   2
#define NUMK    170
#define MTOT    (BATCH * SEQ)   // 2048

// ---------- helpers ----------
__device__ __forceinline__ unsigned short f2bf(float f) {
  unsigned u = __float_as_uint(f);
  u += 0x7FFFu + ((u >> 16) & 1u);          // round-to-nearest-even
  return (unsigned short)(u >> 16);
}
__device__ __forceinline__ unsigned pack2bf(float lo, float hi) {
  return (unsigned)f2bf(lo) | ((unsigned)f2bf(hi) << 16);
}

union FragU8 { unsigned u[8]; v16bf v; };

// A-fragment (16x32 bf16) half-index for element e, lane-group g:
//   V0..3 : K = 2e   + 8g        V4..7 : K = 2e+8 + 8g
__device__ __forceinline__ int a_hk(int e, int g) {
  return (e < 4) ? (2 * e + 8 * g) : (2 * e + 8 + 8 * g);
}
// B-fragment (32x16 bf16): same K order as A (V0..3 : K = 2e + 8g, V4..7 : K = 2e+8 + 8g)
__device__ __forceinline__ int b_hk(int e, int g) { return a_hk(e, g); }

// ================= GEMM:  C[M,N] = A[M,K] * W[N,K]^T + bias =================
// M=2048, N=K=1024 for all four uses. Block tile 64x128, 8 waves, K-step 32,
// double-buffered LDS staging (one barrier per K-step) + L2 prefetch of i+2.
// mode 0: bf16 out [B,H,S,dh]   (Q,K heads)
// mode 1: bf16 out [B,H,dh,S]   (V transposed)
// mode 2: f32  out [M,N]        (final projection)
#define BM 64
#define BN 128
#define BK 32
#define LSTR 36   // LDS row stride in halves (72 B, 8B-aligned)
#define NKSTEP (D_MODEL / BK)   // 32

__device__ __forceinline__ void stage_tiles(const float* __restrict__ A,
                                            const float* __restrict__ W,
                                            int mBase, int nBase, int k0, int tid,
                                            unsigned short* lA, unsigned short* lW) {
  for (int t = tid; t < BM * 8; t += 256) {
    const int r = t >> 3, c4 = (t & 7) << 2;
    const float4 f = *(const float4*)(A + (size_t)(mBase + r) * D_MODEL + k0 + c4);
    *(unsigned*)(lA + r * LSTR + c4)     = pack2bf(f.x, f.y);
    *(unsigned*)(lA + r * LSTR + c4 + 2) = pack2bf(f.z, f.w);
  }
  for (int t = tid; t < BN * 8; t += 256) {
    const int r = t >> 3, c4 = (t & 7) << 2;
    const float4 f = *(const float4*)(W + (size_t)(nBase + r) * D_MODEL + k0 + c4);
    *(unsigned*)(lW + r * LSTR + c4)     = pack2bf(f.x, f.y);
    *(unsigned*)(lW + r * LSTR + c4 + 2) = pack2bf(f.z, f.w);
  }
}

__global__ __launch_bounds__(256)
void gemm_bias_kernel(const float* __restrict__ A, const float* __restrict__ W,
                      const float* __restrict__ bias, void* __restrict__ outp,
                      int mode) {
  __shared__ unsigned short lA[2][BM * LSTR];   // 2 x 4608 B
  __shared__ unsigned short lW[2][BN * LSTR];   // 2 x 9216 B
  const int tid  = threadIdx.x;
  const int lane = tid & 31, wave = tid >> 5;
  const int wm = wave & 3, wn = wave >> 2;     // 4 (M) x 2 (N) wave grid
  const int g = lane >> 4, ln = lane & 15;
  const int mBase = blockIdx.y * BM;
  const int nBase = blockIdx.x * BN;

  v8f acc[4] = {};

  stage_tiles(A, W, mBase, nBase, 0, tid, lA[0], lW[0]);
  __syncthreads();

  for (int i = 0; i < NKSTEP; ++i) {
    const int cur = i & 1;

    if (i + 2 < NKSTEP) {       // L2/L0 prefetch of tile i+2 (global_prefetch_b8)
      const int kp = (i + 2) * BK;
      for (int t = tid; t < BM * 8; t += 256)
        __builtin_prefetch(A + (size_t)(mBase + (t >> 3)) * D_MODEL + kp + ((t & 7) << 2), 0, 3);
      for (int t = tid; t < BN * 8; t += 256)
        __builtin_prefetch(W + (size_t)(nBase + (t >> 3)) * D_MODEL + kp + ((t & 7) << 2), 0, 3);
    }
    if (i + 1 < NKSTEP)         // stage next tile into the alternate buffer
      stage_tiles(A, W, mBase, nBase, (i + 1) * BK, tid, lA[cur ^ 1], lW[cur ^ 1]);

    // compute on current buffer
    FragU8 af;
    const int arow = wm * 16 + ln;
#pragma unroll
    for (int e = 0; e < 8; ++e)
      af.u[e] = *(const unsigned*)(lA[cur] + arow * LSTR + a_hk(e, g));

#pragma unroll
    for (int nt = 0; nt < 4; ++nt) {
      FragU8 bfr;
      const int nrow = wn * 64 + nt * 16 + ln;
#pragma unroll
      for (int e = 0; e < 8; ++e)
        bfr.u[e] = *(const unsigned*)(lW[cur] + nrow * LSTR + b_hk(e, g));
      acc[nt] = __builtin_amdgcn_wmma_f32_16x16x32_bf16(
          false, af.v, false, bfr.v, (short)0, acc[nt], false, false);
    }
    __syncthreads();   // staging of i+1 complete; compute on cur done
  }

  // write out (C layout: element e -> M = e + 8g, N = ln)
#pragma unroll
  for (int nt = 0; nt < 4; ++nt) {
    const int Ng = nBase + wn * 64 + nt * 16 + ln;
    const float bv = bias[Ng];
#pragma unroll
    for (int e = 0; e < 8; ++e) {
      const int Mg = mBase + wm * 16 + e + 8 * g;
      const float val = acc[nt][e] + bv;
      if (mode == 2) {
        ((float*)outp)[(size_t)Mg * D_MODEL + Ng] = val;
      } else {
        const int b = Mg >> 10, s = Mg & 1023;
        const int h = Ng >> 6,  d = Ng & 63;
        const size_t off = (mode == 0)
            ? (((size_t)(b * N_HEAD + h) * SEQ + s) * DH + d)   // [B,H,S,dh]
            : (((size_t)(b * N_HEAD + h) * DH + d) * SEQ + s);  // [B,H,dh,S]
        ((unsigned short*)outp)[off] = f2bf(val);
      }
    }
  }
}

// ================= fused scores + exact top-k softmax + P*V =================
// One WG (128 threads / 4 waves) per (b, h, 16-row q block).
__global__ __launch_bounds__(128)
void attn_kernel(const unsigned short* __restrict__ qh,
                 const unsigned short* __restrict__ kh,
                 const unsigned short* __restrict__ vhT,
                 float* __restrict__ ctx) {
  __shared__ float sc[16 * SEQ];   // 64 KB score / prob slab
  const int tid  = threadIdx.x;
  const int lane = tid & 31, wave = tid >> 5;
  const int g = lane >> 4, ln = lane & 15;
  const int qb = blockIdx.x;           // q block (16 rows)
  const int h  = blockIdx.y;
  const int b  = blockIdx.z;
  const int bh = b * N_HEAD + h;
  const size_t base = (size_t)bh * SEQ * DH;

  // ---- Phase 1: scores = (Qh Kh^T) / 8  -> LDS
  FragU8 a0, a1;
  {
    const unsigned short* qrow = qh + base + (size_t)(qb * 16 + ln) * DH;
#pragma unroll
    for (int e = 0; e < 8; ++e) {
      a0.u[e] = *(const unsigned*)(qrow + a_hk(e, g));
      a1.u[e] = *(const unsigned*)(qrow + 32 + a_hk(e, g));
    }
  }
  for (int kt = 0; kt < 16; ++kt) {            // each wave covers 256 keys
    const int kb = wave * 256 + kt * 16;
    if (kt + 1 < 16)                           // prefetch next key-tile row
      __builtin_prefetch(kh + base + (size_t)(kb + 16 + ln) * DH, 0, 3);
    FragU8 b0, b1;
    const unsigned short* krow = kh + base + (size_t)(kb + ln) * DH;
#pragma unroll
    for (int e = 0; e < 8; ++e) {
      b0.u[e] = *(const unsigned*)(krow + b_hk(e, g));
      b1.u[e] = *(const unsigned*)(krow + 32 + b_hk(e, g));
    }
    v8f c = {};
    c = __builtin_amdgcn_wmma_f32_16x16x32_bf16(false, a0.v, false, b0.v, (short)0, c, false, false);
    c = __builtin_amdgcn_wmma_f32_16x16x32_bf16(false, a1.v, false, b1.v, (short)0, c, false, false);
#pragma unroll
    for (int e = 0; e < 8; ++e)
      sc[(e + 8 * g) * SEQ + kb + ln] = c[e] * 0.125f;
  }
  __syncthreads();

  // ---- Phase 2/3: exact 170th-largest (bitwise radix select) + softmax
  for (int j = 0; j < 4; ++j) {
    const int r = wave * 4 + j;
    float fv[32]; unsigned keys[32];
    float m = -3.4e38f;
#pragma unroll
    for (int i = 0; i < 32; ++i) {
      const float f = sc[r * SEQ + lane + 32 * i];
      fv[i] = f;
      m = fmaxf(m, f);
      const unsigned u = __float_as_uint(f);
      keys[i] = (u & 0x80000000u) ? ~u : (u | 0x80000000u);  // monotone map
    }
#pragma unroll
    for (int o = 16; o > 0; o >>= 1) m = fmaxf(m, __shfl_xor(m, o, 32));

    unsigned prefix = 0u, hm = 0u;
    int kk = NUMK;
    for (int bit = 31; bit >= 0; --bit) {
      const unsigned bmask = 1u << bit;
      int cnt = 0;
#pragma unroll
      for (int i = 0; i < 32; ++i)
        cnt += (((keys[i] & hm) == prefix) && (keys[i] & bmask)) ? 1 : 0;
#pragma unroll
      for (int o = 16; o > 0; o >>= 1) cnt += __shfl_xor(cnt, o, 32);
      if (cnt >= kk) prefix |= bmask; else kk -= cnt;
      hm |= bmask;
    }
    // prefix == key of the 170th-largest score; keep key >= prefix
    float ssum = 0.f;
#pragma unroll
    for (int i = 0; i < 32; ++i)
      if (keys[i] >= prefix) ssum += __expf(fv[i] - m);
#pragma unroll
    for (int o = 16; o > 0; o >>= 1) ssum += __shfl_xor(ssum, o, 32);
    const float inv = 1.0f / ssum;
#pragma unroll
    for (int i = 0; i < 32; ++i) {
      const float p = (keys[i] >= prefix) ? __expf(fv[i] - m) * inv : 0.0f;
      sc[r * SEQ + lane + 32 * i] = p;
    }
  }
  __syncthreads();

  // ---- Phase 4: ctx = P @ Vh  (wave owns 16 of the 64 head dims)
  const int dbase = wave * 16;
  v8f acc = {};
  const unsigned short* vrow = vhT + base + (size_t)(dbase + ln) * SEQ;
  for (int ks = 0; ks < SEQ; ks += 32) {
    if (ks + 32 < SEQ)
      __builtin_prefetch(vrow + ks + 32, 0, 3);
    FragU8 af, bfr;
#pragma unroll
    for (int e = 0; e < 8; ++e) {
      const int col = ks + a_hk(e, g);
      const float2 f = *(const float2*)(sc + ln * SEQ + col);
      af.u[e] = pack2bf(f.x, f.y);
    }
#pragma unroll
    for (int e = 0; e < 8; ++e)
      bfr.u[e] = *(const unsigned*)(vrow + ks + b_hk(e, g));
    acc = __builtin_amdgcn_wmma_f32_16x16x32_bf16(
        false, af.v, false, bfr.v, (short)0, acc, false, false);
  }
#pragma unroll
  for (int e = 0; e < 8; ++e) {
    const int q = qb * 16 + e + 8 * g;
    const int n = h * DH + dbase + ln;
    ctx[(size_t)(b * SEQ + q) * D_MODEL + n] = acc[e];   // [B,S,D]
  }
}

// ============================ launcher ============================
extern "C" void kernel_launch(void* const* d_in, const int* in_sizes, int n_in,
                              void* d_out, int out_size, void* d_ws, size_t ws_size,
                              hipStream_t stream) {
  (void)in_sizes; (void)n_in; (void)out_size; (void)ws_size;
  const float* q  = (const float*)d_in[0];
  const float* k  = (const float*)d_in[1];
  const float* v  = (const float*)d_in[2];
  const float* wq = (const float*)d_in[3];
  const float* bq = (const float*)d_in[4];
  const float* wk = (const float*)d_in[5];
  const float* bk = (const float*)d_in[6];
  const float* wv = (const float*)d_in[7];
  const float* bv = (const float*)d_in[8];
  const float* wo = (const float*)d_in[9];
  const float* bo = (const float*)d_in[10];

  const size_t headElems = (size_t)BATCH * N_HEAD * SEQ * DH;  // 2,097,152
  unsigned short* qh  = (unsigned short*)d_ws;                 // 4 MB bf16
  unsigned short* kh  = qh + headElems;                        // 4 MB bf16
  unsigned short* vhT = kh + headElems;                        // 4 MB bf16
  float* ctx = (float*)(vhT + headElems);                      // 8 MB f32

  dim3 ggrid(D_MODEL / BN, MTOT / BM);   // (8, 32)
  gemm_bias_kernel<<<ggrid, 256, 0, stream>>>(q, wq, bq, (void*)qh, 0);
  gemm_bias_kernel<<<ggrid, 256, 0, stream>>>(k, wk, bk, (void*)kh, 0);
  gemm_bias_kernel<<<ggrid, 256, 0, stream>>>(v, wv, bv, (void*)vhT, 1);

  dim3 agrid(SEQ / 16, N_HEAD, BATCH);   // (64, 16, 2)
  attn_kernel<<<agrid, 128, 0, stream>>>(qh, kh, vhT, ctx);

  gemm_bias_kernel<<<ggrid, 256, 0, stream>>>(ctx, wo, bo, d_out, 2);
}
